// Decoder_50766513439240
// MI455X (gfx1250) — matrix operand order repaired and checked by reading, not compiled, not run on hardware
//
#include <hip/hip_runtime.h>
#include <hip/hip_bf16.h>
#include <stdint.h>

// ---------------------------------------------------------------------------
// MI455X (gfx1250) fused pointer-network decoder step.
// - Big GEMMs (Wa@static + Wb@dynamic, Wd1@static) on v_wmma_f32_16x16x32_bf16,
//   fp32 accumulate. Weights pre-converted to bf16 once, pre-swizzled into
//   WMMA A-fragment order (zero conversion VALU in the hot loop).
// - Encoder tiles staged HBM->LDS by the Tensor Data Mover (tensor_load_to_lds,
//   TENSORcnt), double-buffered, then converted/swizzled to bf16 B fragments.
// - tanh + v-dot epilogue fused (hardware v_tanh_f32), reduced via ds_add_f32,
//   so the [B,H,S] energy tensors are never materialized.
// ---------------------------------------------------------------------------

#define Bb 256
#define Hh 512
#define Ss 512

typedef __attribute__((ext_vector_type(16))) __bf16 v16bf;
typedef __attribute__((ext_vector_type(8)))  float  v8f;
typedef __attribute__((ext_vector_type(4)))  unsigned int v4u;
typedef __attribute__((ext_vector_type(8)))  int v8i_t;
typedef __attribute__((ext_vector_type(4)))  int v4i_t;

union pk2bf {
    __bf16 h[2];
    unsigned u;
};

// ---- hardware tanh (TRANS op on gfx1250) ----------------------------------
#if defined(__has_builtin)
#if __has_builtin(__builtin_amdgcn_tanhf)
#define TANHF(x) __builtin_amdgcn_tanhf(x)
#elif __has_builtin(__builtin_amdgcn_tanh_f32)
#define TANHF(x) __builtin_amdgcn_tanh_f32(x)
#endif
#endif
#ifndef TANHF
__device__ __forceinline__ float hw_tanhf(float x) {
    float y;
    // v_nop honors the TRANS co-execution hazard (1 indep op after trans)
    asm volatile("v_tanh_f32 %0, %1\n\tv_nop" : "=v"(y) : "v"(x));
    return y;
}
#define TANHF(x) hw_tanhf(x)
#endif

// ---- Tensor Data Mover: 2D tile (128 rows x 64 cols of fp32) -> LDS -------
#if defined(__has_builtin)
#if __has_builtin(__builtin_amdgcn_tensor_load_to_lds)
#define HAVE_TDM 1
#endif
#endif

__device__ __forceinline__ unsigned lds_off(const void* p) {
    return (unsigned)(uintptr_t)p;   // low 32 bits of flat LDS address = offset
}

#ifdef HAVE_TDM
__device__ __forceinline__ void tdm_load_tile128x64(unsigned ldsaddr,
                                                    const float* src) {
    // D# group0: count=1 (user, valid), lds_addr, global_addr[56:0], type=2
    v4u g0;
    const uint64_t ga = (uint64_t)(uintptr_t)src;
    g0[0] = 1u;
    g0[1] = ldsaddr;
    g0[2] = (unsigned)ga;
    g0[3] = ((unsigned)(ga >> 32) & 0x01FFFFFFu) | 0x80000000u;  // type=2
    // D# group1: data_size=4B; tensor_dim0=512, tensor_dim1=512,
    //            tile_dim0=64, tile_dim1=128, tensor_dim0_stride=512
    v8i_t g1;
    g1[0] = 0x00020000;          // wg_mask=0 | data_size=2 (4B)
    g1[1] = (int)(512u << 16);   // atomic_barrier_addr=0 | tensor_dim0.lo16
    g1[2] = (int)(512u << 16);   // tensor_dim0.hi16=0    | tensor_dim1.lo16
    g1[3] = (int)(64u << 16);    // tensor_dim1.hi16=0    | tile_dim0=64
    g1[4] = 128;                 // tile_dim1=128 | tile_dim2=0
    g1[5] = 512;                 // tensor_dim0_stride.lo32
    g1[6] = 0;                   // stride.hi16 | tensor_dim1_stride.lo16
    g1[7] = 0;
    v4i_t g2 = {0, 0, 0, 0};     // 2D tensor: groups 2/3 unused
    v4i_t g3 = {0, 0, 0, 0};
#if __clang_major__ >= 23
    v8i_t g4 = {0, 0, 0, 0, 0, 0, 0, 0};
    __builtin_amdgcn_tensor_load_to_lds(g0, g1, g2, g3, g4, 0);
#else
    __builtin_amdgcn_tensor_load_to_lds(g0, g1, g2, g3, 0);
#endif
}
#endif

// ---------------------------------------------------------------------------
// Pre-swizzle one fp32 weight matrix (row-major, pitch/colofs) into bf16
// WMMA A-fragment order:
//   fragment (ht, ks) = rows [ht*16,+16) x K [ks*32,+32)
//   ushort index = (ht*16 + ks)*512 + lane*16 + j
//   lane = (row&15) + 16*((k>>4)&1) ; j = k&15
// ---------------------------------------------------------------------------
__global__ __launch_bounds__(256) void wconv_kernel(
    const float* __restrict__ W, int pitch, int colofs,
    unsigned short* __restrict__ dst)
{
    const int idx = blockIdx.x * 256 + threadIdx.x;   // [0, 512*256)
    const int row = idx >> 8;
    const int k   = (idx & 255) * 2;                  // even
    const float f0 = W[(size_t)row * pitch + colofs + k];
    const float f1 = W[(size_t)row * pitch + colofs + k + 1];
    const int lane = (row & 15) + (((k >> 4) & 1) << 4);
    const int j    = k & 15;                          // even
    const int oi   = ((row >> 4) * 16 + (k >> 5)) * 512 + lane * 16 + j;
    pk2bf p;
    p.h[0] = (__bf16)f0;
    p.h[1] = (__bf16)f1;
    *(unsigned*)&dst[oi] = p.u;
}

// ---------------------------------------------------------------------------
// energy_kernel<HAS_DYN>:
//   out[b,s] = sum_h vvec[h] * tanh( (W1 @ stat[b])[h,s]
//                                  + (W2 @ dyn[b])[h,s]   (if HAS_DYN)
//                                  + addvec[b,h] )
// block = (s-tile of 64, batch b). 256 threads = 8 waves; wave owns 4 M-tiles.
// ---------------------------------------------------------------------------
template <int HAS_DYN>
__global__ __launch_bounds__(256) void energy_kernel(
    const float* __restrict__ stat,            // [B,H,S]
    const float* __restrict__ dyn,             // [B,H,S] or nullptr
    const unsigned short* __restrict__ W1bf,   // pre-swizzled A fragments
    const unsigned short* __restrict__ W2bf,   // or nullptr
    const float* __restrict__ addvec,          // [B,H]
    const float* __restrict__ vvec,            // [H]
    float* __restrict__ out)                   // [B,S]
{
    __shared__ unsigned short fragS[16 * 4 * 512];            // 64 KB
    __shared__ unsigned short fragD[HAS_DYN ? 16 * 4 * 512 : 16];
    __shared__ float chunk[2][128 * 64];                      // 64 KB dbl-buf
    __shared__ float uacc[64];

    const int tid = threadIdx.x;
    const int b   = blockIdx.y;
    const int s0  = blockIdx.x * 64;

    if (tid < 64) uacc[tid] = 0.0f;

    // ---- stage: TDM (HBM -> LDS fp32 chunk) then convert to swizzled bf16 ----
    const int NCH = HAS_DYN ? 8 : 4;   // chunks of 128 k-rows, interleaved S/D
    auto chunk_src = [&](int c) -> const float* {
        const int mat = HAS_DYN ? (c & 1) : 0;
        const int kc  = HAS_DYN ? (c >> 1) : c;
        const float* base = mat ? dyn : stat;
        return base + (size_t)b * Hh * Ss + (size_t)(kc * 128) * Ss + s0;
    };

#ifdef HAVE_TDM
    if (tid < 32) tdm_load_tile128x64(lds_off(&chunk[0][0]), chunk_src(0));
#endif
    for (int c = 0; c < NCH; ++c) {
#ifdef HAVE_TDM
        if (tid < 32) {
            if (c + 1 < NCH) {
                tdm_load_tile128x64(lds_off(&chunk[(c + 1) & 1][0]),
                                    chunk_src(c + 1));
                __builtin_amdgcn_s_wait_tensorcnt(1);   // chunk c complete
            } else {
                __builtin_amdgcn_s_wait_tensorcnt(0);
            }
        }
#else
        {   // fallback: cooperative vector loads
            const float* src = chunk_src(c);
            for (int it = 0; it < 32; ++it) {
                const int p = tid + it * 256;             // 8192 elements
                chunk[c & 1][p] = src[(size_t)(p >> 6) * Ss + (p & 63)];
            }
        }
#endif
        __syncthreads();   // chunk c visible to whole workgroup
        const float* cb = chunk[c & 1];
        unsigned short* dstf = (HAS_DYN && (c & 1)) ? fragD : fragS;
        const int kc = HAS_DYN ? (c >> 1) : c;
#pragma unroll 4
        for (int it = 0; it < 16; ++it) {
            const int p  = tid + it * 256;   // pair index: 64 kpairs x 64 cols
            const int sl = p & 63;
            const int kr = (p >> 6) * 2;     // even chunk row
            const int k  = kc * 128 + kr;
            const int lane = (sl & 15) + ((k & 16) ? 16 : 0);
            const int j    = k & 15;         // even
            const int idx  = ((k >> 5) * 4 + (sl >> 4)) * 512 + lane * 16 + j;
            pk2bf pk;
            pk.h[0] = (__bf16)cb[kr * 64 + sl];
            pk.h[1] = (__bf16)cb[(kr + 1) * 64 + sl];
            *(unsigned*)&dstf[idx] = pk.u;
        }
        __syncthreads();   // buffer (c&1) free for reuse by chunk c+2
    }

    const int wave  = tid >> 5;
    const int lane  = tid & 31;
    const int lrow  = lane & 15;
    const int lhalf = (lane & 16) ? 8 : 0;

    const v16bf* w1f = (const v16bf*)W1bf;   // [ht*16+ks][32 lanes]
    const v16bf* w2f = HAS_DYN ? (const v16bf*)W2bf : nullptr;

#pragma unroll 1
    for (int i = 0; i < 4; ++i) {
        const int ht = wave + 8 * i;           // M-tile index
        const int h0 = ht * 16;
        v8f cacc[4] = {v8f{}, v8f{}, v8f{}, v8f{}};

#pragma unroll 1
        for (int ks = 0; ks < 16; ++ks) {
            const int fi = (ht * 16 + ks) * 32 + lane;
            __builtin_prefetch(&w1f[fi + 32], 0, 0);   // global_prefetch_b8
            v16bf aS = w1f[fi];                        // 2x global_load_b128
            v16bf aD;
            if constexpr (HAS_DYN) aD = w2f[fi];
#pragma unroll
            for (int ns = 0; ns < 4; ++ns) {
                v16bf bS = *(const v16bf*)&fragS[(ks * 4 + ns) * 512 + lane * 16];
                cacc[ns] = __builtin_amdgcn_wmma_f32_16x16x32_bf16(
                    false, aS, false, bS, (short)0, cacc[ns], false, false);
                if constexpr (HAS_DYN) {
                    v16bf bD = *(const v16bf*)&fragD[(ks * 4 + ns) * 512 + lane * 16];
                    cacc[ns] = __builtin_amdgcn_wmma_f32_16x16x32_bf16(
                        false, aD, false, bD, (short)0, cacc[ns], false, false);
                }
            }
        }
        // ---- fused epilogue: +bias-vec, hardware tanh, v-dot reduction ----
        float addv[8], vv[8];
#pragma unroll
        for (int r = 0; r < 8; ++r) {
            const int hidx = h0 + lhalf + r;   // C layout: VGPR r -> M = r (+8 hi)
            addv[r] = addvec[b * Hh + hidx];
            vv[r]   = vvec[hidx];
        }
#pragma unroll
        for (int ns = 0; ns < 4; ++ns) {
            float local = 0.0f;
#pragma unroll
            for (int r = 0; r < 8; ++r)
                local = fmaf(vv[r], TANHF(cacc[ns][r] + addv[r]), local);
            atomicAdd(&uacc[ns * 16 + lrow], local);   // ds_add_f32
        }
    }
    __syncthreads();
    if (tid < 64) out[(size_t)b * Ss + s0 + tid] = uacc[tid];
}

// ---------------------------------------------------------------------------
// embedding: x[b,k] = emb_W[k,0]*o[b,0] + emb_W[k,1]*o[b,1] + emb_b[k]
// ---------------------------------------------------------------------------
__global__ __launch_bounds__(256) void embed_kernel(
    const float* __restrict__ lo, const float* __restrict__ W,
    const float* __restrict__ bias, float* __restrict__ x)
{
    const int i = blockIdx.x * 256 + threadIdx.x;   // B*H
    const int b = i >> 9, k = i & 511;
    x[i] = fmaf(W[k * 2], lo[b * 2],
           fmaf(W[k * 2 + 1], lo[b * 2 + 1], bias[k]));
}

// ---------------------------------------------------------------------------
// single GRU step: block owns 4 hidden indices, stages its 12 weight rows
// (r/z/n for ih+hh) in LDS once, streams all 256 batches.
// ---------------------------------------------------------------------------
__global__ __launch_bounds__(256) void gru_kernel(
    const float* __restrict__ x, const float* __restrict__ hprev,
    const float* __restrict__ Wih, const float* __restrict__ Whh,
    const float* __restrict__ bih, const float* __restrict__ bhh,
    float* __restrict__ hnew)
{
    __shared__ float wih_s[12 * 512];
    __shared__ float whh_s[12 * 512];
    __shared__ float acc_a[12], acc_b[12];

    const int tid = threadIdx.x;
    const int h0  = blockIdx.x * 4;
    for (int i = tid; i < 12 * 512; i += 256) {
        const int lr = i >> 9, k = i & 511;
        const int grow = (lr >> 2) * 512 + h0 + (lr & 3);   // gate*512 + h
        wih_s[i] = Wih[(size_t)grow * 512 + k];
        whh_s[i] = Whh[(size_t)grow * 512 + k];
    }
    __syncthreads();

    const int r = tid >> 4, li = tid & 15;
    for (int b = 0; b < Bb; ++b) {
        float pa = 0.0f, pb = 0.0f;
        if (r < 12) {
            const float* wi = &wih_s[r * 512 + li * 32];
            const float* wh = &whh_s[r * 512 + li * 32];
            const float* xv = x + b * 512 + li * 32;
            const float* hv = hprev + b * 512 + li * 32;
#pragma unroll 8
            for (int k = 0; k < 32; ++k) {
                pa = fmaf(wi[k], xv[k], pa);
                pb = fmaf(wh[k], hv[k], pb);
            }
        }
        for (int m = 8; m >= 1; m >>= 1) {
            pa += __shfl_xor(pa, m, 16);
            pb += __shfl_xor(pb, m, 16);
        }
        if (r < 12 && li == 0) { acc_a[r] = pa; acc_b[r] = pb; }
        __syncthreads();
        if (tid < 4) {
            const int h = h0 + tid;
            const float xr = acc_a[tid]     + bih[h];
            const float hr = acc_b[tid]     + bhh[h];
            const float xz = acc_a[4 + tid] + bih[512 + h];
            const float hz = acc_b[4 + tid] + bhh[512 + h];
            const float xn = acc_a[8 + tid] + bih[1024 + h];
            const float hn = acc_b[8 + tid] + bhh[1024 + h];
            const float rg = 1.0f / (1.0f + __expf(-(xr + hr)));
            const float zg = 1.0f / (1.0f + __expf(-(xz + hz)));
            const float ng = TANHF(xn + rg * hn);
            hnew[b * 512 + h] = (1.0f - zg) * ng + zg * hprev[b * 512 + h];
        }
        __syncthreads();
    }
}

// ---------------------------------------------------------------------------
// out[b, h0+r] = sum_k W[h0+r, colofs+k] * vec[b,k]   (16 rows/block in LDS)
// used for c1 = h_new @ Wc^T
// ---------------------------------------------------------------------------
__global__ __launch_bounds__(256) void rowgemm_kernel(
    const float* __restrict__ W, int pitch, int colofs,
    const float* __restrict__ vec, float* __restrict__ out)
{
    __shared__ float wsm[16 * 512];
    const int tid = threadIdx.x;
    const int h0  = blockIdx.x * 16;
    for (int i = tid; i < 16 * 512; i += 256)
        wsm[i] = W[(size_t)(h0 + (i >> 9)) * pitch + colofs + (i & 511)];
    __syncthreads();

    const int r = tid >> 4, li = tid & 15;
    for (int b = 0; b < Bb; ++b) {
        const float* wrow = &wsm[r * 512 + li * 32];
        const float* v = vec + b * 512 + li * 32;
        float p = 0.0f;
#pragma unroll 8
        for (int k = 0; k < 32; ++k) p = fmaf(wrow[k], v[k], p);
        for (int m = 8; m >= 1; m >>= 1) p += __shfl_xor(p, m, 16);
        if (li == 0) out[b * 512 + h0 + r] = p;
    }
}

// ---------------------------------------------------------------------------
// per-batch: softmax(logits) -> context = attn @ static^T -> c2 = Wd2 @ context
// ---------------------------------------------------------------------------
__global__ __launch_bounds__(256) void softmax_ctx_kernel(
    const float* __restrict__ logits, const float* __restrict__ stat,
    const float* __restrict__ decW, float* __restrict__ c2out)
{
    __shared__ float att[512];
    __shared__ float ctx[512];
    __shared__ float red[8];

    const int tid = threadIdx.x, w = tid >> 5, lane = tid & 31;
    const int b = blockIdx.x;

    const float a0 = logits[b * 512 + tid];
    const float a1 = logits[b * 512 + 256 + tid];

    float m = fmaxf(a0, a1);
    for (int s = 16; s >= 1; s >>= 1) m = fmaxf(m, __shfl_xor(m, s, 32));
    if (lane == 0) red[w] = m;
    __syncthreads();
    m = red[0];
#pragma unroll
    for (int i = 1; i < 8; ++i) m = fmaxf(m, red[i]);

    const float e0 = __expf(a0 - m), e1 = __expf(a1 - m);
    float ssum = e0 + e1;
    for (int s = 16; s >= 1; s >>= 1) ssum += __shfl_xor(ssum, s, 32);
    __syncthreads();
    if (lane == 0) red[w] = ssum;
    __syncthreads();
    float tot = 0.0f;
#pragma unroll
    for (int i = 0; i < 8; ++i) tot += red[i];
    const float inv = 1.0f / tot;
    att[tid] = e0 * inv;
    att[tid + 256] = e1 * inv;
    __syncthreads();

    const float* sb = stat + (size_t)b * Hh * Ss;
    for (int it = 0; it < 64; ++it) {
        const int h = it * 8 + w;
        const float* row = sb + (size_t)h * Ss;
        float sum = 0.0f;
#pragma unroll 4
        for (int i = 0; i < 16; ++i) {
            const int s = lane + 32 * i;
            sum = fmaf(att[s], row[s], sum);
        }
        for (int s = 16; s >= 1; s >>= 1) sum += __shfl_xor(sum, s, 32);
        if (lane == 0) ctx[h] = sum;
    }
    __syncthreads();

    for (int it = 0; it < 64; ++it) {
        const int h = it * 8 + w;
        const float* wrow = decW + (size_t)h * 1024 + 512;   // Wd2 = dec_W[:, H:2H]
        float sum = 0.0f;
#pragma unroll 4
        for (int i = 0; i < 16; ++i) {
            const int k = lane + 32 * i;
            sum = fmaf(wrow[k], ctx[k], sum);
        }
        for (int s = 16; s >= 1; s >>= 1) sum += __shfl_xor(sum, s, 32);
        if (lane == 0) c2out[b * 512 + h] = sum;
    }
}

// ---------------------------------------------------------------------------
extern "C" void kernel_launch(void* const* d_in, const int* in_sizes, int n_in,
                              void* d_out, int out_size, void* d_ws, size_t ws_size,
                              hipStream_t stream) {
    const float* static_enc  = (const float*)d_in[0];
    const float* dynamic_enc = (const float*)d_in[1];
    const float* last_output = (const float*)d_in[2];
    const float* last_hidden = (const float*)d_in[3];
    const float* emb_W       = (const float*)d_in[4];
    const float* emb_b       = (const float*)d_in[5];
    const float* gru_Wih     = (const float*)d_in[6];
    const float* gru_Whh     = (const float*)d_in[7];
    const float* gru_bih     = (const float*)d_in[8];
    const float* gru_bhh     = (const float*)d_in[9];
    const float* attn_v      = (const float*)d_in[10];
    const float* attn_W      = (const float*)d_in[11];
    const float* dec_v       = (const float*)d_in[12];
    const float* dec_W       = (const float*)d_in[13];

    float* out  = (float*)d_out;          // [B,S] pointer logits
    float* hnew = out + Bb * Ss;          // [1,B,H] hidden output

    float* ws     = (float*)d_ws;
    float* x      = ws;                   // [B,H]
    float* c1     = ws + 1 * Bb * Hh;     // [B,H]  Wc @ h_new
    float* logits = ws + 2 * Bb * Hh;     // [B,S]  attention logits
    float* c2     = ws + 3 * Bb * Hh;     // [B,H]  Wd2 @ context
    unsigned short* waBF  = (unsigned short*)(ws + 4 * Bb * Hh);   // 512KB each
    unsigned short* wbBF  = waBF + Hh * Hh;
    unsigned short* wd1BF = wbBF + Hh * Hh;

    // weight convert + swizzle into WMMA A-fragment order
    wconv_kernel<<<512, 256, 0, stream>>>(attn_W, 3 * Hh, 0,  waBF);
    wconv_kernel<<<512, 256, 0, stream>>>(attn_W, 3 * Hh, Hh, wbBF);
    wconv_kernel<<<512, 256, 0, stream>>>(dec_W,  2 * Hh, 0,  wd1BF);

    embed_kernel<<<Bb * Hh / 256, 256, 0, stream>>>(last_output, emb_W, emb_b, x);
    gru_kernel<<<Hh / 4, 256, 0, stream>>>(x, last_hidden, gru_Wih, gru_Whh,
                                           gru_bih, gru_bhh, hnew);
    rowgemm_kernel<<<Hh / 16, 256, 0, stream>>>(attn_W, 3 * Hh, 2 * Hh, hnew, c1);
    energy_kernel<1><<<dim3(Ss / 64, Bb), 256, 0, stream>>>(
        static_enc, dynamic_enc, waBF, wbBF, c1, attn_v, logits);
    softmax_ctx_kernel<<<Bb, 256, 0, stream>>>(logits, static_enc, dec_W, c2);
    energy_kernel<0><<<dim3(Ss / 64, Bb), 256, 0, stream>>>(
        static_enc, nullptr, wd1BF, nullptr, c2, dec_v, out);
}
